// ViTAttention_86895778333017
// MI455X (gfx1250) — matrix operand order repaired and checked by reading, not compiled, not run on hardware
//
#include <hip/hip_runtime.h>
#include <hip/hip_bf16.h>

// ---------------------------------------------------------------------------
// ViT attention for MI455X (gfx1250): bf16 WMMA + TDM (tensor_load_to_lds)
// B=16, N=1024, H=1024, NH=16, HD=64.
// ---------------------------------------------------------------------------

typedef __attribute__((ext_vector_type(16))) __bf16 v16bf;
typedef __attribute__((ext_vector_type(8)))  __bf16 v8bf;
typedef __attribute__((ext_vector_type(8)))  float  v8f;
typedef __attribute__((ext_vector_type(4)))  unsigned int v4u;
typedef __attribute__((ext_vector_type(4)))  int v4i;
typedef __attribute__((ext_vector_type(8)))  int v8i_t;

#define NTOK 1024
#define HDIM 1024
#define NHEAD 16
#define HD 64
#define BATCH 16
#define BN (BATCH * NTOK)   // 16384

#if __has_builtin(__builtin_amdgcn_tensor_load_to_lds) && \
    __has_builtin(__builtin_amdgcn_s_wait_tensorcnt)
#define USE_TDM 1
#else
#define USE_TDM 0
#endif

__device__ __forceinline__ v8f wmma_bf16(v16bf a, v16bf b, v8f c) {
  return __builtin_amdgcn_wmma_f32_16x16x32_bf16(
      /*neg_a=*/false, a, /*neg_b=*/false, b,
      /*c_mod=*/(short)0, c, /*reuse_a=*/false, /*reuse_b=*/false);
}

// A-matrix fragment (16x32, row major, leading dim ld), per ISA VGPR layout.
__device__ __forceinline__ v16bf load_frag_a(const __bf16* p, int ld) {
  int lane = threadIdx.x & 31;
  int row  = lane & 15;
  int kb   = (lane >> 4) << 3;           // 0 or 8
  const __bf16* q = p + row * ld + kb;
  v8bf lo = *(const v8bf*)(q);
  v8bf hi = *(const v8bf*)(q + 16);
  v16bf r;
#pragma unroll
  for (int i = 0; i < 8; ++i) { r[i] = lo[i]; r[i + 8] = hi[i]; }
  return r;
}

// B-matrix fragment (32x16); source stored transposed pt[n][k], ld = K stride.
__device__ __forceinline__ v16bf load_frag_b(const __bf16* pt, int ld) {
  int lane = threadIdx.x & 31;
  int col  = lane & 15;
  int kb   = (lane >> 4) << 3;           // 0 or 8
  const __bf16* q = pt + col * ld + kb;
  v8bf lo = *(const v8bf*)(q);
  v8bf hi = *(const v8bf*)(q + 16);
  v16bf r;
#pragma unroll
  for (int i = 0; i < 8; ++i) { r[i] = lo[i]; r[i + 8] = hi[i]; }
  return r;
}

// ---------------------------------------------------------------------------
// TDM: async 2-D tile load Global -> LDS. D# bitfields per CDNA5 ISA §8.
// ---------------------------------------------------------------------------
__device__ __forceinline__ void tdm_load_2d(unsigned lds_off, const void* gptr,
                                            unsigned tensor_d0, unsigned tensor_d1,
                                            unsigned tile_d0, unsigned tile_d1,
                                            unsigned stride0_elts) {
#if USE_TDM
  unsigned long long ga = (unsigned long long)gptr;
  v4u g0;
  g0.x = 1u;                                        // count=1, user mode
  g0.y = lds_off;                                   // lds_addr (bytes)
  g0.z = (unsigned)ga;                              // global_addr[31:0]
  g0.w = (unsigned)((ga >> 32) & 0x1FFFFFFu)        // global_addr[56:32]
         | 0x80000000u;                             // type=2 ("image")
  v8i_t g1;
  g1[0] = (int)(1u << 16);                          // data_size=1 -> 2 bytes
  g1[1] = (int)((tensor_d0 & 0xFFFFu) << 16);       // tensor_dim0[15:0]
  g1[2] = (int)(((tensor_d0 >> 16) & 0xFFFFu) |     // tensor_dim0[31:16]
                ((tensor_d1 & 0xFFFFu) << 16));     // tensor_dim1[15:0]
  g1[3] = (int)(((tensor_d1 >> 16) & 0xFFFFu) |     // tensor_dim1[31:16]
                (tile_d0 << 16));                   // tile_dim0
  g1[4] = (int)tile_d1;                             // tile_dim1 (tile_dim2=0)
  g1[5] = (int)stride0_elts;                        // tensor_dim0_stride[31:0]
  g1[6] = 0;
  g1[7] = 0;
  v4i z = {0, 0, 0, 0};
#if defined(__clang_major__) && (__clang_major__ >= 23)
  v8i_t z8;
#pragma unroll
  for (int i = 0; i < 8; ++i) z8[i] = 0;
  __builtin_amdgcn_tensor_load_to_lds(g0, g1, z, z, z8, 0);
#else
  __builtin_amdgcn_tensor_load_to_lds(g0, g1, z, z, 0);
#endif
#endif
}

__device__ __forceinline__ unsigned lds_off_of(const void* p) {
  // Generic (flat) shared address: addr[31:0] is the LDS byte offset.
  return (unsigned)(unsigned long long)p;
}

// ---------------------------------------------------------------------------
// Conversions
// ---------------------------------------------------------------------------
__global__ __launch_bounds__(256) void cvt_x_k(const float* __restrict__ x,
                                               __bf16* __restrict__ xb, int n) {
  int i = blockIdx.x * 256 + threadIdx.x;
  if (i < n) xb[i] = (__bf16)x[i];
}

__global__ __launch_bounds__(256) void cvt_wT_k(const float* __restrict__ w,
                                                __bf16* __restrict__ wt) {
  int i = blockIdx.x * 256 + threadIdx.x;   // 1M elements
  int n = i >> 10;
  int k = i & 1023;
  wt[i] = (__bf16)w[k * HDIM + n];          // wt[n][k] = w[k][n]
}

// ---------------------------------------------------------------------------
// GEMM: Y[BN x H] = Xb @ W (+ bias), W transposed bf16, TDM-fed LDS tiles.
// MODE 0: Q bf16 [B,NH,N,HD] * 0.125  MODE 1: K bf16 [B,NH,N,HD]
// MODE 2: V bf16 [B,NH,HD,N]          MODE 3: fp32 [BN,H]
// Block 256 thr = 8 waves; tile 128x128; wave 32x64 (2x4 WMMA); K chunks 32.
// ---------------------------------------------------------------------------
template <int MODE>
__device__ __forceinline__ void store_tile(const v8f& c, int m0, int n0,
                                           void* outp, float scale) {
  int lane  = threadIdx.x & 31;
  int colL  = lane & 15;
  int rbase = (lane >> 4) << 3;
#pragma unroll
  for (int v = 0; v < 8; ++v) {
    int rg = m0 + rbase + v;
    int cg = n0 + colL;
    float val = c[v];
    if (MODE == 3) {
      ((float*)outp)[(size_t)rg * HDIM + cg] = val;
    } else {
      int b = rg >> 10, n = rg & 1023;
      int h = cg >> 6,  d = cg & 63;
      __bf16 bv = (__bf16)(val * scale);
      if (MODE == 2)
        ((__bf16*)outp)[(size_t)(((b * NHEAD + h) * HD + d)) * NTOK + n] = bv;
      else
        ((__bf16*)outp)[(size_t)(((b * NHEAD + h) * NTOK + n)) * HD + d] = bv;
    }
  }
}

template <int MODE>
__global__ __launch_bounds__(256) void gemm_bf16_k(
    const __bf16* __restrict__ A, const __bf16* __restrict__ Bt,
    const float* __restrict__ bias, void* __restrict__ outp, float scale) {
  __shared__ __bf16 sA[2][128 * 32];     // 8 KB per buffer
  __shared__ __bf16 sB[2][128 * 32];     // 8 KB per buffer

  int tid  = threadIdx.x;
  int wave = tid >> 5;
  int lane = tid & 31;
  int wrow = (wave >> 1) * 32;           // 0,32,64,96
  int wcol = (wave & 1) * 64;            // 0,64
  int m0 = blockIdx.x * 128;
  int n0 = blockIdx.y * 128;

  v8f c[2][4];
#pragma unroll
  for (int j = 0; j < 4; ++j) {
    float bcol = bias[n0 + wcol + j * 16 + (lane & 15)];
#pragma unroll
    for (int v = 0; v < 8; ++v) { c[0][j][v] = bcol; c[1][j][v] = bcol; }
  }

  const __bf16* Abase = A + (size_t)m0 * HDIM;
  const __bf16* Bbase = Bt + (size_t)n0 * HDIM;
  int wv0 = __builtin_amdgcn_readfirstlane((int)(tid >> 5));

#if USE_TDM
  // -------- async double-buffered pipeline: TDM overlaps WMMA --------
  if (wv0 == 0) {
    tdm_load_2d(lds_off_of(&sA[0][0]), Abase, HDIM, BN,   32, 128, HDIM);
    tdm_load_2d(lds_off_of(&sB[0][0]), Bbase, HDIM, HDIM, 32, 128, HDIM);
  }
  for (int kc = 0; kc < HDIM; kc += 32) {
    int cur = (kc >> 5) & 1;
    if (wv0 == 0) __builtin_amdgcn_s_wait_tensorcnt((short)0);
    __syncthreads();                     // buffer `cur` ready for all waves
    if (kc + 32 < HDIM && wv0 == 0) {
      tdm_load_2d(lds_off_of(&sA[cur ^ 1][0]), Abase + kc + 32, HDIM, BN,   32, 128, HDIM);
      tdm_load_2d(lds_off_of(&sB[cur ^ 1][0]), Bbase + kc + 32, HDIM, HDIM, 32, 128, HDIM);
    }
    const __bf16* sa = &sA[cur][0] + wrow * 32;
    v16bf a0 = load_frag_a(sa, 32);
    v16bf a1 = load_frag_a(sa + 16 * 32, 32);
#pragma unroll
    for (int j = 0; j < 4; ++j) {
      v16bf bj = load_frag_b(&sB[cur][0] + (wcol + j * 16) * 32, 32);
      c[0][j] = wmma_bf16(a0, bj, c[0][j]);
      c[1][j] = wmma_bf16(a1, bj, c[1][j]);
    }
    __syncthreads();                     // all reads of `cur` done
  }
#else
  // -------- fallback: cooperative vector loads into LDS --------
  for (int kc = 0; kc < HDIM; kc += 32) {
#pragma unroll
    for (int i = 0; i < 2; ++i) {
      int idx = tid + i * 256;           // 0..511 -> 128 rows x 4 chunks
      int row = idx >> 2;
      int c8  = (idx & 3) * 8;
      *(uint4*)&sA[0][row * 32 + c8] = *(const uint4*)&Abase[(size_t)row * HDIM + kc + c8];
      *(uint4*)&sB[0][row * 32 + c8] = *(const uint4*)&Bbase[(size_t)row * HDIM + kc + c8];
    }
    __syncthreads();
    const __bf16* sa = &sA[0][0] + wrow * 32;
    v16bf a0 = load_frag_a(sa, 32);
    v16bf a1 = load_frag_a(sa + 16 * 32, 32);
#pragma unroll
    for (int j = 0; j < 4; ++j) {
      v16bf bj = load_frag_b(&sB[0][0] + (wcol + j * 16) * 32, 32);
      c[0][j] = wmma_bf16(a0, bj, c[0][j]);
      c[1][j] = wmma_bf16(a1, bj, c[1][j]);
    }
    __syncthreads();
  }
#endif

#pragma unroll
  for (int i = 0; i < 2; ++i)
#pragma unroll
    for (int j = 0; j < 4; ++j)
      store_tile<MODE>(c[i][j], m0 + wrow + i * 16, n0 + wcol + j * 16, outp, scale);
}

// ---------------------------------------------------------------------------
// Flash attention: one wave = 16-query tile of one (b,h). Q pre-scaled 0.125.
// Q,K: bf16 [B,NH,N,HD]; Vt: bf16 [B,NH,HD,N]; O: bf16 [B,N,H].
// ---------------------------------------------------------------------------
__global__ __launch_bounds__(128) void flash_attn_k(
    const __bf16* __restrict__ Q, const __bf16* __restrict__ K,
    const __bf16* __restrict__ Vt, __bf16* __restrict__ O) {
  __shared__ __bf16 lp[4][16 * 32];      // per-wave P staging (D->A transpose)
  int wave = threadIdx.x >> 5;
  int lane = threadIdx.x & 31;
  int gw = blockIdx.x * 4 + wave;
  int bh = gw >> 6;
  int q0 = (gw & 63) << 4;

  const __bf16* Qb = Q + ((size_t)bh * NTOK + q0) * HD;
  const __bf16* Kb = K + (size_t)bh * NTOK * HD;
  const __bf16* Vb = Vt + (size_t)bh * HD * NTOK;
  __bf16* lw = lp[wave];

  v16bf qa0 = load_frag_a(Qb, HD);
  v16bf qa1 = load_frag_a(Qb + 32, HD);

  float mrow[8], lrow[8];
  v8f o0, o1, o2, o3;
#pragma unroll
  for (int v = 0; v < 8; ++v) {
    mrow[v] = -1e30f; lrow[v] = 0.0f;
    o0[v] = 0.0f; o1[v] = 0.0f; o2[v] = 0.0f; o3[v] = 0.0f;
  }

  for (int kc = 0; kc < NTOK; kc += 32) {
    v16bf kb00 = load_frag_b(Kb + (size_t)kc * HD, HD);
    v16bf kb01 = load_frag_b(Kb + (size_t)kc * HD + 32, HD);
    v8f s0;
#pragma unroll
    for (int v = 0; v < 8; ++v) s0[v] = 0.0f;
    s0 = wmma_bf16(qa0, kb00, s0);
    s0 = wmma_bf16(qa1, kb01, s0);

    v16bf kb10 = load_frag_b(Kb + (size_t)(kc + 16) * HD, HD);
    v16bf kb11 = load_frag_b(Kb + (size_t)(kc + 16) * HD + 32, HD);
    v8f s1;
#pragma unroll
    for (int v = 0; v < 8; ++v) s1[v] = 0.0f;
    s1 = wmma_bf16(qa0, kb10, s1);
    s1 = wmma_bf16(qa1, kb11, s1);

#pragma unroll
    for (int v = 0; v < 8; ++v) {
      float mx = fmaxf(s0[v], s1[v]);
      mx = fmaxf(mx, __shfl_xor(mx, 1));
      mx = fmaxf(mx, __shfl_xor(mx, 2));
      mx = fmaxf(mx, __shfl_xor(mx, 4));
      mx = fmaxf(mx, __shfl_xor(mx, 8));
      mx = fmaxf(mx, mrow[v]);
      float alpha = __expf(mrow[v] - mx);
      float p0 = __expf(s0[v] - mx);
      float p1 = __expf(s1[v] - mx);
      float rs = p0 + p1;
      rs += __shfl_xor(rs, 1);
      rs += __shfl_xor(rs, 2);
      rs += __shfl_xor(rs, 4);
      rs += __shfl_xor(rs, 8);
      lrow[v] = lrow[v] * alpha + rs;
      mrow[v] = mx;
      o0[v] *= alpha; o1[v] *= alpha; o2[v] *= alpha; o3[v] *= alpha;
      int r = v + ((lane >> 4) << 3);
      lw[r * 32 + (lane & 15)]      = (__bf16)p0;
      lw[r * 32 + 16 + (lane & 15)] = (__bf16)p1;
    }

    v16bf pa = load_frag_a(lw, 32);
    v16bf vb0 = load_frag_b(Vb + 0  * NTOK + kc, NTOK);
    v16bf vb1 = load_frag_b(Vb + 16 * NTOK + kc, NTOK);
    v16bf vb2 = load_frag_b(Vb + 32 * NTOK + kc, NTOK);
    v16bf vb3 = load_frag_b(Vb + 48 * NTOK + kc, NTOK);
    o0 = wmma_bf16(pa, vb0, o0);
    o1 = wmma_bf16(pa, vb1, o1);
    o2 = wmma_bf16(pa, vb2, o2);
    o3 = wmma_bf16(pa, vb3, o3);
  }

  int b = bh >> 4, h = bh & 15;
#pragma unroll
  for (int v = 0; v < 8; ++v) {
    float inv = 1.0f / lrow[v];
    int r = v + ((lane >> 4) << 3);
    size_t base = (size_t)(b * NTOK + q0 + r) * HDIM + h * HD + (lane & 15);
    O[base + 0]  = (__bf16)(o0[v] * inv);
    O[base + 16] = (__bf16)(o1[v] * inv);
    O[base + 32] = (__bf16)(o2[v] * inv);
    O[base + 48] = (__bf16)(o3[v] * inv);
  }
}

// ---------------------------------------------------------------------------
// Launch
// ---------------------------------------------------------------------------
extern "C" void kernel_launch(void* const* d_in, const int* in_sizes, int n_in,
                              void* d_out, int out_size, void* d_ws, size_t ws_size,
                              hipStream_t stream) {
  const float* x  = (const float*)d_in[0];
  const float* wq = (const float*)d_in[1];
  const float* bq = (const float*)d_in[2];
  const float* wk = (const float*)d_in[3];
  const float* bk = (const float*)d_in[4];
  const float* wv = (const float*)d_in[5];
  const float* bv = (const float*)d_in[6];
  const float* wo = (const float*)d_in[7];
  const float* bo = (const float*)d_in[8];
  float* out = (float*)d_out;

  char* ws = (char*)d_ws;
  size_t off = 0;
  auto alloc = [&](size_t bytes) {
    void* p = ws + off;
    off += (bytes + 255) & ~(size_t)255;
    return p;
  };
  __bf16* xb  = (__bf16*)alloc((size_t)BN * HDIM * 2);
  __bf16* wqT = (__bf16*)alloc((size_t)HDIM * HDIM * 2);
  __bf16* wkT = (__bf16*)alloc((size_t)HDIM * HDIM * 2);
  __bf16* wvT = (__bf16*)alloc((size_t)HDIM * HDIM * 2);
  __bf16* woT = (__bf16*)alloc((size_t)HDIM * HDIM * 2);
  __bf16* Qb  = (__bf16*)alloc((size_t)BN * HDIM * 2);
  __bf16* Kb  = (__bf16*)alloc((size_t)BN * HDIM * 2);
  __bf16* Vt  = (__bf16*)alloc((size_t)BN * HDIM * 2);
  __bf16* Ob  = (__bf16*)alloc((size_t)BN * HDIM * 2);

  cvt_x_k<<<(BN * HDIM) / 256, 256, 0, stream>>>(x, xb, BN * HDIM);
  cvt_wT_k<<<(HDIM * HDIM) / 256, 256, 0, stream>>>(wq, wqT);
  cvt_wT_k<<<(HDIM * HDIM) / 256, 256, 0, stream>>>(wk, wkT);
  cvt_wT_k<<<(HDIM * HDIM) / 256, 256, 0, stream>>>(wv, wvT);
  cvt_wT_k<<<(HDIM * HDIM) / 256, 256, 0, stream>>>(wo, woT);

  dim3 gg(BN / 128, HDIM / 128);
  gemm_bf16_k<0><<<gg, 256, 0, stream>>>(xb, wqT, bq, Qb, 0.125f);
  gemm_bf16_k<1><<<gg, 256, 0, stream>>>(xb, wkT, bk, Kb, 1.0f);
  gemm_bf16_k<2><<<gg, 256, 0, stream>>>(xb, wvT, bv, Vt, 1.0f);

  flash_attn_k<<<(BATCH * NHEAD * (NTOK / 16)) / 4, 128, 0, stream>>>(Qb, Kb, Vt, Ob);

  gemm_bf16_k<3><<<gg, 256, 0, stream>>>(Ob, woT, bo, out, 1.0f);
}
